// AttentionBlock_15642270892311
// MI455X (gfx1250) — matrix operand order repaired and checked by reading, not compiled, not run on hardware
//
#include <hip/hip_runtime.h>

// ---------------------------------------------------------------------------
// MI455X (gfx1250, wave32) AttentionBlock: GN -> QKV -> softmax(QK^T)V -> proj
// All GEMMs on v_wmma_f32_16x16x32_bf16 (fp32 accumulate). Intermediates bf16
// (fits in the 192MB L2). Attention fused flash-style: no score tensor in HBM.
// ---------------------------------------------------------------------------

typedef __attribute__((ext_vector_type(16))) __bf16      v16bf;
typedef __attribute__((ext_vector_type(8)))  float       v8f;
typedef __attribute__((ext_vector_type(4)))  unsigned int u32x4;

union FragBF {           // one WMMA bf16 A/B operand (16 halves = 32B)
  v16bf v;
  u32x4 q[2];
};
union Pack8 {            // 8 packed bf16 for one 16B vector store
  u32x4  q;
  __bf16 h[8];
};

constexpr int BATCH = 32;
constexpr int CH    = 512;
constexpr int HW    = 1024;          // 32*32 tokens
constexpr int NGRP  = 8;
constexpr int CPG   = CH / NGRP;     // 64 channels per group

#define WMMA_BF16(a, b, c)                                                     \
  __builtin_amdgcn_wmma_f32_16x16x32_bf16(false, (a), false, (b), (short)0,    \
                                          (c), false, false)

// ---------------------------------------------------------------------------
// K0: GroupNorm statistics. One workgroup per (batch, group); groups tile the
// channel dim contiguously so the reduction range is one flat 64*1024 slab.
// ---------------------------------------------------------------------------
__global__ __launch_bounds__(256) void gn_stats_kernel(
    const float* __restrict__ x, float* __restrict__ stats) {
  const int bg = blockIdx.x;                       // 0..255 == b*NGRP + g
  const float* p = x + (size_t)bg * CPG * HW;
  const int n = CPG * HW;                          // 65536
  float s = 0.f, s2 = 0.f;
  for (int i = threadIdx.x; i < n; i += 256) {
    float v = p[i];
    s += v;
    s2 += v * v;
  }
  __shared__ float ls[256], ls2[256];
  ls[threadIdx.x] = s;
  ls2[threadIdx.x] = s2;
  __syncthreads();
  for (int off = 128; off > 0; off >>= 1) {
    if (threadIdx.x < off) {
      ls[threadIdx.x] += ls[threadIdx.x + off];
      ls2[threadIdx.x] += ls2[threadIdx.x + off];
    }
    __syncthreads();
  }
  if (threadIdx.x == 0) {
    float mu  = ls[0] / (float)n;
    float var = ls2[0] / (float)n - mu * mu;       // biased var (jnp default)
    stats[2 * bg + 0] = mu;
    stats[2 * bg + 1] = rsqrtf(var + 1e-5f);
  }
}

// ---------------------------------------------------------------------------
// K1: convert GEMM weights fp32 -> bf16 once (operand precision for WMMA).
// ---------------------------------------------------------------------------
__global__ __launch_bounds__(256) void cvt_w_kernel(
    const float* __restrict__ qkvw, const float* __restrict__ projw,
    __bf16* __restrict__ qkvw_bf, __bf16* __restrict__ projw_bf) {
  const int i = blockIdx.x * 256 + threadIdx.x;
  if (i < 3 * CH * CH) qkvw_bf[i] = (__bf16)qkvw[i];
  if (i < CH * CH)     projw_bf[i] = (__bf16)projw[i];
}

// ---------------------------------------------------------------------------
// K2: fused GroupNorm-apply + QKV GEMM.
//   D[oc, token] = sum_c W[oc,c] * h[c,token],  h = gn(x)
// Workgroup = 128 threads (4 waves), 64x64 output tile, K-step 32.
// h tile is normalized on load and staged transposed in LDS ([token][c],
// row stride 40 halves -> conflict-free 16B reads for B fragments).
// Outputs: q,k token-major [B,HW,C] bf16; v channel-major [B,C,HW] bf16
// (layouts chosen so every downstream WMMA fragment load is contiguous).
// ---------------------------------------------------------------------------
__global__ __launch_bounds__(128) void qkv_gemm_kernel(
    const float* __restrict__ x, const float* __restrict__ stats,
    const float* __restrict__ gamma, const float* __restrict__ beta,
    const __bf16* __restrict__ wbf, const float* __restrict__ bias,
    __bf16* __restrict__ q_t, __bf16* __restrict__ k_t,
    __bf16* __restrict__ v_t) {
  const int b    = blockIdx.z;
  const int m0   = blockIdx.y * 64;          // output channel tile (0..1472)
  const int n0   = blockIdx.x * 64;          // token tile
  const int tid  = threadIdx.x;
  const int wave = tid >> 5;
  const int lane = tid & 31;
  const int l16  = lane & 15;
  const int half = lane >> 4;
  const int mw   = m0 + 16 * wave;           // this wave's 16 output rows

  __shared__ __align__(16) __bf16 hT[64][40]; // [token][c], pad 8 halves

  const float* xb = x + (size_t)b * CH * HW;

  const v8f vzero = {0.f, 0.f, 0.f, 0.f, 0.f, 0.f, 0.f, 0.f};
  v8f acc[4];
#pragma unroll
  for (int t = 0; t < 4; ++t) acc[t] = vzero;

  const int cl  = tid >> 2;                  // 0..31: channel within K-step
  const int tk0 = (tid & 3) * 16;            // token sub-chunk

  for (int c0 = 0; c0 < CH; c0 += 32) {
    { // stage normalized x^T tile: 32 channels x 64 tokens
      const int c = c0 + cl;
      const int g = c / CPG;
      const float mu = stats[2 * (b * NGRP + g) + 0];
      const float rs = stats[2 * (b * NGRP + g) + 1];
      const float ga = gamma[c] * rs;
      const float be = beta[c] - mu * ga;    // h = x*ga + be
      const float* src = xb + (size_t)c * HW + n0 + tk0;
#pragma unroll
      for (int i = 0; i < 16; ++i)
        hT[tk0 + i][cl] = (__bf16)(src[i] * ga + be);
    }
    __syncthreads();

    // A fragment: 16x32 tile of W (row-major, contiguous 16B chunks)
    FragBF a;
    const __bf16* wrow = wbf + (size_t)(mw + l16) * CH + c0;
    a.q[0] = *(const u32x4*)(wrow + 16 * half);
    a.q[1] = *(const u32x4*)(wrow + 16 * half + 8);

#pragma unroll
    for (int nt = 0; nt < 4; ++nt) {
      FragBF bm;  // B fragment: lane = token, K = 16*half + t (contiguous)
      const __bf16* hrow = &hT[nt * 16 + l16][16 * half];
      bm.q[0] = *(const u32x4*)(hrow);
      bm.q[1] = *(const u32x4*)(hrow + 8);
      acc[nt] = WMMA_BF16(a.v, bm.v, acc[nt]);
    }
    __syncthreads();
  }

  float bv[8];
#pragma unroll
  for (int r = 0; r < 8; ++r) bv[r] = bias[mw + 8 * half + r];

  const int region = mw >> 9;                // 0=q, 1=k, 2=v (512-aligned)
  const int ocb    = (mw & 511) + 8 * half;  // multiple of 8 -> 16B aligned
#pragma unroll
  for (int nt = 0; nt < 4; ++nt) {
    const int token = n0 + nt * 16 + l16;
    if (region < 2) {                        // q/k: token-major, vector store
      __bf16* dst =
          (region == 0 ? q_t : k_t) + ((size_t)b * HW + token) * CH + ocb;
      Pack8 pk;
#pragma unroll
      for (int r = 0; r < 8; ++r) pk.h[r] = (__bf16)(acc[nt][r] + bv[r]);
      *(u32x4*)dst = pk.q;
    } else {                                 // v: channel-major
      __bf16* dst = v_t + ((size_t)b * CH + ocb) * HW + token;
#pragma unroll
      for (int r = 0; r < 8; ++r)
        dst[(size_t)r * HW] = (__bf16)(acc[nt][r] + bv[r]);
    }
  }
}

// ---------------------------------------------------------------------------
// K3: fused attention. One workgroup (4 waves) per (batch, 16-query block).
// Phase A: S = Q K^T; each wave owns 256 keys -> 16 f32 accum tiles (128
// VGPRs/lane). Safe softmax via cross-wave LDS reduction, probabilities to
// LDS bf16. Phase B: O = P V; each wave owns 128 channels (8 accum tiles),
// divide by row-sum at the end. Score matrix never touches HBM.
// ---------------------------------------------------------------------------
__global__ __launch_bounds__(128) void attn_kernel(
    const __bf16* __restrict__ q_t, const __bf16* __restrict__ k_t,
    const __bf16* __restrict__ v_t, __bf16* __restrict__ o_t) {
  const int b    = blockIdx.y;
  const int qt0  = blockIdx.x * 16;          // query token base
  const int tid  = threadIdx.x;
  const int wave = tid >> 5;
  const int lane = tid & 31;
  const int l16  = lane & 15;
  const int half = lane >> 4;

  __shared__ __align__(16) __bf16 P[16][1032];   // probs, pad 8 halves
  __shared__ float red[128][8];                  // cross-wave reduce scratch
  __shared__ float rowstat[16];                  // row max, then row sum

  const __bf16* qb = q_t + ((size_t)b * HW + qt0) * CH;
  const __bf16* kb = k_t + (size_t)b * HW * CH;
  const __bf16* vb = v_t + (size_t)b * CH * HW;

  const v8f vzero = {0.f, 0.f, 0.f, 0.f, 0.f, 0.f, 0.f, 0.f};
  v8f s[16];
#pragma unroll
  for (int t = 0; t < 16; ++t) s[t] = vzero;

  const int key0 = wave * 256;

  // ---- Phase A: logits -------------------------------------------------
  for (int c0 = 0; c0 < CH; c0 += 32) {
    FragBF a;                                // Q rows (shared across tiles)
    const __bf16* qrow = qb + (size_t)l16 * CH + c0;
    a.q[0] = *(const u32x4*)(qrow + 16 * half);
    a.q[1] = *(const u32x4*)(qrow + 16 * half + 8);
#pragma unroll
    for (int kt = 0; kt < 16; ++kt) {
      FragBF bm;                             // K^T: lane = key, K = channel
      const __bf16* krow =
          kb + (size_t)(key0 + kt * 16 + l16) * CH + c0 + 16 * half;
      bm.q[0] = *(const u32x4*)(krow);
      bm.q[1] = *(const u32x4*)(krow + 8);
      s[kt] = WMMA_BF16(a.v, bm.v, s[kt]);
    }
  }

  // ---- softmax: row max ------------------------------------------------
  float pm[8];
#pragma unroll
  for (int r = 0; r < 8; ++r) {
    float m = s[0][r];
#pragma unroll
    for (int kt = 1; kt < 16; ++kt) m = fmaxf(m, s[kt][r]);
    pm[r] = m;
  }
#pragma unroll
  for (int r = 0; r < 8; ++r) red[tid][r] = pm[r];
  __syncthreads();
  if (tid < 16) {                            // one thread per query row
    const int hh = tid >> 3, r = tid & 7;
    float m = -3.0e38f;
    for (int w = 0; w < 4; ++w)
      for (int l = 0; l < 16; ++l) m = fmaxf(m, red[w * 32 + hh * 16 + l][r]);
    rowstat[tid] = m;
  }
  __syncthreads();
  float rmax[8];
#pragma unroll
  for (int r = 0; r < 8; ++r) rmax[r] = rowstat[8 * half + r];

  // ---- exp + stage P in LDS + partial sums -----------------------------
  const float k2 = 0.044194173824159216f * 1.4426950408889634f; // scale*log2e
  float psum[8];
#pragma unroll
  for (int r = 0; r < 8; ++r) psum[r] = 0.f;
#pragma unroll
  for (int kt = 0; kt < 16; ++kt) {
#pragma unroll
    for (int r = 0; r < 8; ++r) {
      float p = exp2f((s[kt][r] - rmax[r]) * k2);
      psum[r] += p;
      P[8 * half + r][key0 + kt * 16 + l16] = (__bf16)p;
    }
  }
  __syncthreads();                            // rmax reads done; safe to reuse
#pragma unroll
  for (int r = 0; r < 8; ++r) red[tid][r] = psum[r];
  __syncthreads();
  if (tid < 16) {
    const int hh = tid >> 3, r = tid & 7;
    float ssum = 0.f;
    for (int w = 0; w < 4; ++w)
      for (int l = 0; l < 16; ++l) ssum += red[w * 32 + hh * 16 + l][r];
    rowstat[tid] = ssum;                      // now holds row sums
  }
  __syncthreads();                            // P + sums visible to all waves

  // ---- Phase B: O = P * V ---------------------------------------------
  v8f o[8];
#pragma unroll
  for (int ct = 0; ct < 8; ++ct) o[ct] = vzero;

  for (int k0 = 0; k0 < HW; k0 += 32) {
    FragBF a;                                 // P from LDS: M=query, K=key
    const __bf16* prow = &P[l16][k0 + 16 * half];
    a.q[0] = *(const u32x4*)(prow);
    a.q[1] = *(const u32x4*)(prow + 8);
#pragma unroll
    for (int ct = 0; ct < 8; ++ct) {
      const int chan = wave * 128 + ct * 16 + l16;
      FragBF bm;                              // V channel-major: contiguous
      const __bf16* vrow = vb + (size_t)chan * HW + k0 + 16 * half;
      bm.q[0] = *(const u32x4*)(vrow);
      bm.q[1] = *(const u32x4*)(vrow + 8);
      o[ct] = WMMA_BF16(a.v, bm.v, o[ct]);
    }
  }

  float rs[8];
#pragma unroll
  for (int r = 0; r < 8; ++r) rs[r] = 1.0f / rowstat[8 * half + r];
#pragma unroll
  for (int ct = 0; ct < 8; ++ct) {
    const int chan = wave * 128 + ct * 16 + l16;
#pragma unroll
    for (int r = 0; r < 8; ++r) {
      const int token = qt0 + 8 * half + r;
      o_t[((size_t)b * HW + token) * CH + chan] = (__bf16)(o[ct][r] * rs[r]);
    }
  }
}

// ---------------------------------------------------------------------------
// K4: proj GEMM + bias + residual (fp32 out). Same tiling as K2; B fragments
// straight from token-major o_t (contiguous 32B per lane, L2-resident).
// ---------------------------------------------------------------------------
__global__ __launch_bounds__(128) void proj_gemm_kernel(
    const __bf16* __restrict__ wbf, const __bf16* __restrict__ o_t,
    const float* __restrict__ bias, const float* __restrict__ x,
    float* __restrict__ out) {
  const int b    = blockIdx.z;
  const int m0   = blockIdx.y * 64;
  const int n0   = blockIdx.x * 64;
  const int tid  = threadIdx.x;
  const int wave = tid >> 5;
  const int lane = tid & 31;
  const int l16  = lane & 15;
  const int half = lane >> 4;
  const int mw   = m0 + 16 * wave;

  const v8f vzero = {0.f, 0.f, 0.f, 0.f, 0.f, 0.f, 0.f, 0.f};
  v8f acc[4];
#pragma unroll
  for (int t = 0; t < 4; ++t) acc[t] = vzero;

  const __bf16* ob = o_t + (size_t)b * HW * CH;

  for (int c0 = 0; c0 < CH; c0 += 32) {
    FragBF a;
    const __bf16* wrow = wbf + (size_t)(mw + l16) * CH + c0;
    a.q[0] = *(const u32x4*)(wrow + 16 * half);
    a.q[1] = *(const u32x4*)(wrow + 16 * half + 8);
#pragma unroll
    for (int nt = 0; nt < 4; ++nt) {
      FragBF bm;
      const __bf16* orow =
          ob + (size_t)(n0 + nt * 16 + l16) * CH + c0 + 16 * half;
      bm.q[0] = *(const u32x4*)(orow);
      bm.q[1] = *(const u32x4*)(orow + 8);
      acc[nt] = WMMA_BF16(a.v, bm.v, acc[nt]);
    }
  }

  float bv[8];
#pragma unroll
  for (int r = 0; r < 8; ++r) bv[r] = bias[mw + 8 * half + r];
#pragma unroll
  for (int nt = 0; nt < 4; ++nt) {
    const int token = n0 + nt * 16 + l16;
#pragma unroll
    for (int r = 0; r < 8; ++r) {
      const size_t idx = ((size_t)b * CH + mw + 8 * half + r) * HW + token;
      out[idx] = x[idx] + bv[r] + acc[nt][r];
    }
  }
}

// ---------------------------------------------------------------------------
extern "C" void kernel_launch(void* const* d_in, const int* in_sizes, int n_in,
                              void* d_out, int out_size, void* d_ws,
                              size_t ws_size, hipStream_t stream) {
  (void)in_sizes; (void)n_in; (void)out_size; (void)ws_size;
  const float* x      = (const float*)d_in[0];
  const float* norm_w = (const float*)d_in[1];
  const float* norm_b = (const float*)d_in[2];
  const float* qkv_w  = (const float*)d_in[3];
  const float* qkv_b  = (const float*)d_in[4];
  const float* proj_w = (const float*)d_in[5];
  const float* proj_b = (const float*)d_in[6];
  float* out = (float*)d_out;

  char* ws = (char*)d_ws;
  size_t off = 0;
  auto carve = [&](size_t bytes) -> void* {
    void* p = ws + off;
    off = (off + bytes + 255) & ~(size_t)255;
    return p;
  };
  float*  stats    = (float*) carve((size_t)BATCH * NGRP * 2 * sizeof(float));
  __bf16* qkvw_bf  = (__bf16*)carve((size_t)3 * CH * CH * sizeof(__bf16));
  __bf16* projw_bf = (__bf16*)carve((size_t)CH * CH * sizeof(__bf16));
  __bf16* q_t      = (__bf16*)carve((size_t)BATCH * HW * CH * sizeof(__bf16));
  __bf16* k_t      = (__bf16*)carve((size_t)BATCH * HW * CH * sizeof(__bf16));
  __bf16* v_t      = (__bf16*)carve((size_t)BATCH * CH * HW * sizeof(__bf16));
  __bf16* o_t      = (__bf16*)carve((size_t)BATCH * HW * CH * sizeof(__bf16));

  gn_stats_kernel<<<BATCH * NGRP, 256, 0, stream>>>(x, stats);
  cvt_w_kernel<<<(3 * CH * CH + 255) / 256, 256, 0, stream>>>(
      qkv_w, proj_w, qkvw_bf, projw_bf);
  qkv_gemm_kernel<<<dim3(HW / 64, 3 * CH / 64, BATCH), 128, 0, stream>>>(
      x, stats, norm_w, norm_b, qkvw_bf, qkv_b, q_t, k_t, v_t);
  attn_kernel<<<dim3(HW / 16, BATCH), 128, 0, stream>>>(q_t, k_t, v_t, o_t);
  proj_gemm_kernel<<<dim3(HW / 64, CH / 64, BATCH), 128, 0, stream>>>(
      projw_bf, o_t, proj_b, x, out);
}
